// IntegerCifar10Net_66838281060421
// MI455X (gfx1250) — 8-bit matrix operand order repaired and checked by reading, compiled, not run on hardware
//
#include <hip/hip_runtime.h>
#include <cstdint>
#include <cstddef>

typedef __attribute__((ext_vector_type(8))) int v8i;

__device__ __forceinline__ signed char quantize_w(float w) {
  w = fminf(fmaxf(w, -1.f), 1.f);
  return (signed char)(int)rintf(w * 7.f);
}

__global__ void prep_conv_w(const float* __restrict__ w, signed char* __restrict__ qw,
                            int CI, int total) {
  int i = blockIdx.x * blockDim.x + threadIdx.x;
  if (i >= total) return;
  const int k9 = 9 * CI;
  const int co = i / k9;
  const int rem = i - co * k9;
  const int e = rem / CI;
  const int ci = rem - e * CI;
  const int dy = e / 3, dx = e - dy * 3;
  qw[i] = quantize_w(w[(((size_t)co * CI + ci) * 3 + dy) * 3 + dx]);
}

__global__ void prep_fc1_w(const float* __restrict__ w, signed char* __restrict__ qw, int total) {
  int i = blockIdx.x * blockDim.x + threadIdx.x;
  if (i >= total) return;
  const int o = i >> 12;
  const int knew = i & 4095;
  const int c = knew & 255;
  const int sp = knew >> 8;
  qw[i] = quantize_w(w[((size_t)o << 12) + c * 16 + sp]);
}

__global__ void prep_fc2_w(const float* __restrict__ w, signed char* __restrict__ qw, int total) {
  int i = blockIdx.x * blockDim.x + threadIdx.x;
  if (i >= total) return;
  qw[i] = quantize_w(w[i]);
}

__global__ void qconv1_f32(const float* __restrict__ x,
                           const signed char* __restrict__ qw1,
                           const float* __restrict__ gam,
                           const float* __restrict__ bet,
                           unsigned char* __restrict__ out) {
  __shared__ float sx[3][3][34];
  __shared__ float sw[64][27];
  const int y = blockIdx.x & 31;
  const int img = blockIdx.x >> 5;
  for (int i = threadIdx.x; i < 64 * 27; i += blockDim.x)
    sw[i / 27][i % 27] = (float)qw1[i] * (1.f / 7.f);
  for (int i = threadIdx.x; i < 3 * 3 * 34; i += blockDim.x) {
    const int ch = i / 102;
    const int rr = (i - ch * 102) / 34;
    const int cc = i % 34;
    const int gy = y - 1 + rr;
    const int gx = cc - 1;
    float v = 0.f;
    if (gy >= 0 && gy < 32 && gx >= 0 && gx < 32)
      v = x[(((size_t)img * 3 + ch) * 32 + gy) * 32 + gx];
    sx[ch][rr][cc] = v;
  }
  __syncthreads();
  const int xcol = threadIdx.x & 31;
  const int cog = threadIdx.x >> 5;
#pragma unroll
  for (int j = 0; j < 8; ++j) {
    const int co = (cog << 3) + j;
    float s = 0.f;
#pragma unroll
    for (int dy = 0; dy < 3; ++dy)
#pragma unroll
      for (int dx = 0; dx < 3; ++dx)
#pragma unroll
        for (int ci = 0; ci < 3; ++ci)
          s += sx[ci][dy][xcol + dx] * sw[co][(dy * 3 + dx) * 3 + ci];
    float t = s * gam[co] + bet[co];
    t = fminf(fmaxf(t, -1.f), 1.f);
    out[(((size_t)img * 32 + y) * 32 + xcol) * 64 + co] =
        (unsigned char)fmaxf(rintf(t * 7.f), 0.f);
  }
}

template <int CIN>
__global__ void qconv_wmma(const unsigned char* __restrict__ actin,
                           const signed char* __restrict__ qw,
                           const float* __restrict__ gam,
                           const float* __restrict__ bet,
                           unsigned char* __restrict__ actout,
                           int B, int H, int W, int COUT) {
  extern __shared__ __align__(16) unsigned char smem[];
  const int XB = (W >= 16) ? 16 : W;
  const int RW = 16 / XB;
  const int nw = blockDim.x >> 5;
  const int rows_wg = nw * RW;
  const int tile_rows = rows_wg + 2;
  const int tile_cols = XB + 2;
  unsigned char* sa = smem;
  signed char* sw = (signed char*)(smem + (size_t)tile_rows * tile_cols * CIN);

  const int wave = threadIdx.x >> 5;
  const int lane = threadIdx.x & 31;

  const int ct = COUT >> 4;
  const int xt = W / XB;
  const int yt = H / rows_wg;
  int bid = blockIdx.x;
  const int cb = bid % ct; bid /= ct;
  const int xb = bid % xt; bid /= xt;
  const int yb = bid % yt; bid /= yt;
  const int img = bid;
  const int co0 = cb << 4;
  const int x0 = xb * XB;
  const int y0 = yb * rows_wg;

  {
    const int nvec = 16 * 9 * CIN / 16;
    const int4* src = (const int4*)(qw + (size_t)co0 * 9 * CIN);
    int4* dst = (int4*)sw;
    for (int i = threadIdx.x; i < nvec; i += blockDim.x) dst[i] = src[i];
  }
  {
    const int vpc = CIN / 16;
    const int nv = tile_rows * tile_cols * vpc;
    for (int i = threadIdx.x; i < nv; i += blockDim.x) {
      const int cell = i / vpc;
      const int part = i - cell * vpc;
      const int r = cell / tile_cols;
      const int c = cell - r * tile_cols;
      const int gy = y0 - 1 + r;
      const int gx = x0 - 1 + c;
      int4 v = make_int4(0, 0, 0, 0);
      if (gy >= 0 && gy < H && gx >= 0 && gx < W)
        v = *(const int4*)(actin + ((((size_t)img * H + gy) * W) + gx) * CIN + part * 16);
      *(int4*)(sa + (size_t)cell * CIN + part * 16) = v;
    }
  }
  __syncthreads();

  const int m = lane & 15;
  const int half = lane >> 4;
  const int n = lane & 15;
  const int orow = wave * RW + n / XB;
  const int ocol = n % XB;
  const unsigned char* sab = sa + ((size_t)orow * tile_cols + ocol) * CIN;
  const signed char* swm = sw + (size_t)m * 9 * CIN;

  v8i acc = {0, 0, 0, 0, 0, 0, 0, 0};
#pragma unroll
  for (int e = 0; e < 9; ++e) {
    const int dy = e / 3, dx = e - dy * 3;
    const signed char* swe = swm + e * CIN;
    const unsigned char* sae = sab + ((size_t)dy * tile_cols + dx) * CIN;
#pragma unroll
    for (int ci0 = 0; ci0 < CIN; ci0 += 64) {
      v8i a, bf;
#pragma unroll
      for (int v = 0; v < 8; ++v) {
        const int ka = ((v & 4) ? 32 : 0) + ((v & 2) ? 16 : 0) + (half << 3) + ((v & 1) << 2);
        a[v] = *(const int*)(swe + ci0 + ka);
        const int kb = ((v & 4) ? 32 : 0) + ((v & 2) ? 16 : 0) + (half << 3) + ((v & 1) << 2);
        bf[v] = *(const int*)(sae + ci0 + kb);
      }
      acc = __builtin_amdgcn_wmma_i32_16x16x64_iu8(true  , a,
                                                   false  , bf,
                                                   acc, false, false);
    }
  }

  const int gy = y0 + orow;
  const int gx = x0 + ocol;
  unsigned int lo = 0, hi = 0;
#pragma unroll
  for (int r = 0; r < 8; ++r) {
    const int co = co0 + (half << 3) + r;
    float t = (float)acc[r] * (1.f / 49.f);
    t = t * gam[co] + bet[co];
    t = fminf(fmaxf(t, -1.f), 1.f);
    const unsigned int q = (unsigned int)fmaxf(rintf(t * 7.f), 0.f);
    if (r < 4) lo |= q << (r * 8); else hi |= q << ((r - 4) * 8);
  }
  uint2 o2; o2.x = lo; o2.y = hi;
  *(uint2*)(actout + ((((size_t)img * H + gy) * W) + gx) * COUT + co0 + (half << 3)) = o2;
}

__global__ void pool2x2(const unsigned char* __restrict__ in, unsigned char* __restrict__ out,
                        int Ho, int Wo, int C, int total) {
  int i = blockIdx.x * blockDim.x + threadIdx.x;
  if (i >= total) return;
  const int c = i % C;
  int t = i / C;
  const int x = t % Wo; t /= Wo;
  const int y = t % Ho;
  const int b = t / Ho;
  const int W = Wo * 2;
  const size_t base = (((size_t)b * (Ho * 2) + y * 2) * W + x * 2) * C + c;
  const size_t rs = (size_t)W * C;
  unsigned char m0 = in[base], m1 = in[base + C], m2 = in[base + rs], m3 = in[base + rs + C];
  unsigned char a = m0 > m1 ? m0 : m1;
  unsigned char bm = m2 > m3 ? m2 : m3;
  out[i] = a > bm ? a : bm;
}

__global__ void qfc1_wmma(const unsigned char* __restrict__ act,
                          const signed char* __restrict__ qw,
                          const float* __restrict__ gam,
                          const float* __restrict__ bet,
                          unsigned char* __restrict__ out,
                          int ntB, int ntO) {
  const int wave = threadIdx.x >> 5;
  const int lane = threadIdx.x & 31;
  const int tile = blockIdx.x * (blockDim.x >> 5) + wave;
  if (tile >= ntB * ntO) return;
  const int b0 = (tile % ntB) << 4;
  const int o0 = (tile / ntB) << 4;
  const int m = lane & 15, half = lane >> 4, n = lane & 15;
  const signed char* wrow = qw + ((size_t)(o0 + m) << 12);
  const unsigned char* arow = act + ((size_t)(b0 + n) << 12);
  v8i acc = {0, 0, 0, 0, 0, 0, 0, 0};
  for (int k0 = 0; k0 < 4096; k0 += 64) {
    v8i a, bf;
#pragma unroll
    for (int v = 0; v < 8; ++v) {
      const int ka = ((v & 4) ? 32 : 0) + ((v & 2) ? 16 : 0) + (half << 3) + ((v & 1) << 2);
      a[v] = *(const int*)(wrow + k0 + ka);
      const int kb = ((v & 4) ? 32 : 0) + ((v & 2) ? 16 : 0) + (half << 3) + ((v & 1) << 2);
      bf[v] = *(const int*)(arow + k0 + kb);
    }
    acc = __builtin_amdgcn_wmma_i32_16x16x64_iu8(true, a, false, bf, acc, false, false);
  }
  unsigned int lo = 0, hi = 0;
#pragma unroll
  for (int r = 0; r < 8; ++r) {
    const int o = o0 + (half << 3) + r;
    float t = (float)acc[r] * (1.f / 49.f) * gam[o] + bet[o];
    t = fminf(fmaxf(t, -1.f), 1.f);
    const unsigned int q = (unsigned int)fmaxf(rintf(t * 7.f), 0.f);
    if (r < 4) lo |= q << (r * 8); else hi |= q << ((r - 4) * 8);
  }
  uint2 o2; o2.x = lo; o2.y = hi;
  *(uint2*)(out + ((size_t)(b0 + n) << 9) + o0 + (half << 3)) = o2;
}

__global__ void qfc2(const unsigned char* __restrict__ act, const signed char* __restrict__ qw,
                     const float* __restrict__ gam, const float* __restrict__ bet,
                     float* __restrict__ out, int total) {
  int i = blockIdx.x * blockDim.x + threadIdx.x;
  if (i >= total) return;
  const int o = i % 10, b = i / 10;
  const unsigned char* a = act + ((size_t)b << 9);
  const signed char* w = qw + o * 512;
  int s = 0;
  for (int k = 0; k < 512; ++k) s += (int)a[k] * (int)w[k];
  float t = (float)s * (1.f / 49.f) * gam[o] + bet[o];
  t = fminf(fmaxf(t, -1.f), 1.f);
  out[i] = rintf(t * 7.f) * (1.f / 7.f);
}

extern "C" void kernel_launch(void* const* d_in, const int* in_sizes, int n_in,
                              void* d_out, int out_size, void* d_ws, size_t ws_size,
                              hipStream_t stream) {
  (void)n_in; (void)out_size; (void)ws_size;
  const int B = in_sizes[0] / (3 * 32 * 32);

  const float* x   = (const float*)d_in[0];
  const float* w1  = (const float*)d_in[1];  const float* g1  = (const float*)d_in[2];  const float* b1  = (const float*)d_in[3];
  const float* w2  = (const float*)d_in[4];  const float* g2  = (const float*)d_in[5];  const float* b2  = (const float*)d_in[6];
  const float* w3  = (const float*)d_in[7];  const float* g3  = (const float*)d_in[8];  const float* b3  = (const float*)d_in[9];
  const float* w4  = (const float*)d_in[10]; const float* g4  = (const float*)d_in[11]; const float* b4  = (const float*)d_in[12];
  const float* w5  = (const float*)d_in[13]; const float* g5  = (const float*)d_in[14]; const float* b5  = (const float*)d_in[15];
  const float* w6  = (const float*)d_in[16]; const float* g6  = (const float*)d_in[17]; const float* b6  = (const float*)d_in[18];
  const float* wf1 = (const float*)d_in[19]; const float* gf1 = (const float*)d_in[20]; const float* bf1 = (const float*)d_in[21];
  const float* wf2 = (const float*)d_in[22]; const float* gf2 = (const float*)d_in[23]; const float* bf2 = (const float*)d_in[24];

  char* base = (char*)d_ws;
  size_t off = 0;
  auto take = [&](size_t nbytes) -> void* {
    void* r = base + off;
    off += (nbytes + 255) & ~(size_t)255;
    return r;
  };
  signed char* qw1  = (signed char*)take(64 * 27);
  signed char* qw2  = (signed char*)take(64 * 576);
  signed char* qw3  = (signed char*)take((size_t)128 * 576);
  signed char* qw4  = (signed char*)take((size_t)128 * 1152);
  signed char* qw5  = (signed char*)take((size_t)256 * 1152);
  signed char* qw6  = (signed char*)take((size_t)256 * 2304);
  signed char* qwf1 = (signed char*)take((size_t)512 * 4096);
  signed char* qwf2 = (signed char*)take(10 * 512);
  unsigned char* actA = (unsigned char*)take((size_t)B * 32 * 32 * 64);
  unsigned char* actB = (unsigned char*)take((size_t)B * 32 * 32 * 64);

  auto cdiv = [](int a, int b) { return (a + b - 1) / b; };

  prep_conv_w<<<cdiv(64 * 27, 256), 256, 0, stream>>>(w1, qw1, 3, 64 * 27);
  prep_conv_w<<<cdiv(64 * 576, 256), 256, 0, stream>>>(w2, qw2, 64, 64 * 576);
  prep_conv_w<<<cdiv(128 * 576, 256), 256, 0, stream>>>(w3, qw3, 64, 128 * 576);
  prep_conv_w<<<cdiv(128 * 1152, 256), 256, 0, stream>>>(w4, qw4, 128, 128 * 1152);
  prep_conv_w<<<cdiv(256 * 1152, 256), 256, 0, stream>>>(w5, qw5, 128, 256 * 1152);
  prep_conv_w<<<cdiv(256 * 2304, 256), 256, 0, stream>>>(w6, qw6, 256, 256 * 2304);
  prep_fc1_w<<<cdiv(512 * 4096, 256), 256, 0, stream>>>(wf1, qwf1, 512 * 4096);
  prep_fc2_w<<<cdiv(10 * 512, 256), 256, 0, stream>>>(wf2, qwf2, 10 * 512);

  qconv1_f32<<<B * 32, 256, 0, stream>>>(x, qw1, g1, b1, actA);

  auto run_conv = [&](const unsigned char* ain, const signed char* qw, const float* g,
                      const float* bi, unsigned char* aout, int CIN, int H, int W, int COUT) {
    const int XB = (W >= 16) ? 16 : W;
    const int RW = 16 / XB;
    int nw = H / RW; if (nw > 8) nw = 8;
    const int rows_wg = nw * RW;
    const int grid = B * (H / rows_wg) * (W / XB) * (COUT / 16);
    const int block = nw * 32;
    const size_t smem = (size_t)(rows_wg + 2) * (XB + 2) * CIN + (size_t)16 * 9 * CIN;
    if (CIN == 64)
      qconv_wmma<64><<<grid, block, smem, stream>>>(ain, qw, g, bi, aout, B, H, W, COUT);
    else if (CIN == 128)
      qconv_wmma<128><<<grid, block, smem, stream>>>(ain, qw, g, bi, aout, B, H, W, COUT);
    else
      qconv_wmma<256><<<grid, block, smem, stream>>>(ain, qw, g, bi, aout, B, H, W, COUT);
  };
  auto run_pool = [&](const unsigned char* in, unsigned char* out, int Ho, int Wo, int C) {
    const int total = B * Ho * Wo * C;
    pool2x2<<<cdiv(total, 256), 256, 0, stream>>>(in, out, Ho, Wo, C, total);
  };

  run_conv(actA, qw2, g2, b2, actB, 64, 32, 32, 64);
  run_pool(actB, actA, 16, 16, 64);
  run_conv(actA, qw3, g3, b3, actB, 64, 16, 16, 128);
  run_conv(actB, qw4, g4, b4, actA, 128, 16, 16, 128);
  run_pool(actA, actB, 8, 8, 128);
  run_conv(actB, qw5, g5, b5, actA, 128, 8, 8, 256);
  run_conv(actA, qw6, g6, b6, actB, 256, 8, 8, 256);
  run_pool(actB, actA, 4, 4, 256);

  {
    const int ntB = B / 16, ntO = 512 / 16;
    const int tiles = ntB * ntO;
    qfc1_wmma<<<cdiv(tiles, 8), 256, 0, stream>>>(actA, qwf1, gf1, bf1, actB, ntB, ntO);
  }
  {
    const int total = B * 10;
    qfc2<<<cdiv(total, 256), 256, 0, stream>>>(actB, qwf2, gf2, bf2, (float*)d_out, total);
  }
}
